// MultiHeadAttention_51462298141259
// MI455X (gfx1250) — matrix operand order repaired and checked by reading, not compiled, not run on hardware
//
#include <hip/hip_runtime.h>
#include <cstdint>

#define NB  2
#define SEQ 2048
#define DIM 1024
#define NH  16
#define HD  64

typedef __attribute__((ext_vector_type(16))) __bf16 v16bf;
typedef __attribute__((ext_vector_type(8)))  __bf16 v8bf;
typedef __attribute__((ext_vector_type(4)))  __bf16 v4bf;
typedef __attribute__((ext_vector_type(8)))  float  v8f;
typedef __attribute__((ext_vector_type(4)))  float  v4f;
typedef __attribute__((ext_vector_type(4)))  unsigned int u32x4;
typedef __attribute__((ext_vector_type(4)))  int    i32x4;
typedef __attribute__((ext_vector_type(8)))  int    i32x8;

union U16 { v16bf v; v8bf h[2]; };

__device__ inline v8f wmma_bf16(v16bf a, v16bf b, v8f c) {
  // D = A(16x32) * B(32x16) + C, f32 accumulate
  return __builtin_amdgcn_wmma_f32_16x16x32_bf16(
      /*neg_a=*/false, a, /*neg_b=*/false, b,
      /*c_mod=*/(short)0, c, /*reuse_a=*/false, /*reuse_b=*/false);
}

// A-matrix fragment (16x32 bf16): lane = M row; element e -> K = e + 8*grp + (e>=8 ? 8 : 0)
__device__ inline v16bf load_a_frag(const __bf16* rowp, int kk, int grp) {
  U16 u;
  u.h[0] = *reinterpret_cast<const v8bf*>(rowp + kk + 8 * grp);
  u.h[1] = *reinterpret_cast<const v8bf*>(rowp + kk + 8 * grp + 16);
  return u.v;
}

// B-matrix fragment (32x16 bf16, column-major): lane = N col; element e -> K = e + 8*grp + (e>=8 ? 8 : 0)
__device__ inline v16bf load_b_frag(const __bf16* colp, int kk, int grp) {
  U16 u;
  u.h[0] = *reinterpret_cast<const v8bf*>(colp + kk + 8 * grp);
  u.h[1] = *reinterpret_cast<const v8bf*>(colp + kk + 8 * grp + 16);
  return u.v;
}

// ---- Tensor Data Mover: 2D tile (global -> LDS), bf16 elements (data_size code 1) ----
// D# per CDNA5 ISA ch.8: group0 = {count/flags, lds_addr, global_addr lo, global_addr hi|type=2}
// group1 packs data_size, tensor_dim0/1, tile_dim0/1, tensor_dim0_stride. Groups 2/3 zero (2D).
__device__ inline void tdm_load_2d(const void* gptr, unsigned lds_byte_off,
                                   unsigned d0, unsigned d1, unsigned stride_elems) {
  unsigned long long ga = (unsigned long long)(uintptr_t)gptr;
  u32x4 g0;
  g0[0] = 1u;                                     // count=1, user descriptor, no gather
  g0[1] = lds_byte_off;                           // lds_addr (bytes)
  g0[2] = (unsigned)(ga & 0xFFFFFFFFu);           // global_addr[31:0]
  g0[3] = (unsigned)((ga >> 32) & 0x01FFFFFFull)  // global_addr[56:32]
          | (2u << 30);                           // type = 2 ("image")
  i32x8 g1;
  g1[0] = (int)(1u << 16);                        // wg_mask=0, data_size=1 (2 bytes)
  g1[1] = (int)((d0 & 0xFFFFu) << 16);            // tensor_dim0[15:0] -> bits[63:48]
  g1[2] = (int)(((d0 >> 16) & 0xFFFFu) | ((d1 & 0xFFFFu) << 16));  // td0 hi | tensor_dim1 lo
  g1[3] = (int)(((d1 >> 16) & 0xFFFFu) | ((d0 & 0xFFFFu) << 16));  // td1 hi | tile_dim0
  g1[4] = (int)(d1 & 0xFFFFu);                    // tile_dim1 (tile_dim2 = 0)
  g1[5] = (int)stride_elems;                      // tensor_dim0_stride[31:0] (data_size units)
  g1[6] = 0;                                      // stride hi / dim1_stride
  g1[7] = 0;
  i32x4 z4 = {0, 0, 0, 0};
  i32x8 z8 = {0, 0, 0, 0, 0, 0, 0, 0};
  // 6-arg form (clang-23 / therock-10.0 headers)
  __builtin_amdgcn_tensor_load_to_lds(g0, g1, z4, z4, z8, 0);
}

__device__ inline unsigned lds_off(const void* p) {
  return (unsigned)(uintptr_t)p;   // low 32 bits of flat LDS address = byte offset
}

// ---------------- fp32 -> bf16 conversion (vectorized x4) ----------------
__global__ void cvt_bf16_kernel(const float* __restrict__ in, __bf16* __restrict__ out, int n) {
  int i = (blockIdx.x * blockDim.x + threadIdx.x) * 4;
  if (i < n) {
    v4f f = *reinterpret_cast<const v4f*>(in + i);
    v4bf o = __builtin_convertvector(f, v4bf);
    *reinterpret_cast<v4bf*>(out + i) = o;
  }
}

// ---------------- projection GEMM: Y = X @ W^T + b ----------------
// MODE 0: Y bf16 [4096,1024] | MODE 1: Y bf16 [B,H,HD,SEQ] (V transposed) | MODE 2: Y fp32
template <int MODE>
__global__ __launch_bounds__(128) void proj_kernel(const __bf16* __restrict__ X,
                                                   const __bf16* __restrict__ Wm,
                                                   const float* __restrict__ bias,
                                                   void* __restrict__ Yv) {
  const int lane = threadIdx.x & 31;
  const int l16  = lane & 15;
  const int grp  = lane >> 4;
  const int w    = threadIdx.x >> 5;
  const int rowbase = blockIdx.y * 64 + (w >> 1) * 32;
  const int colbase = blockIdx.x * 64 + (w & 1) * 32;

  const __bf16* a0p = X  + (size_t)(rowbase + l16)      * DIM;
  const __bf16* a1p = X  + (size_t)(rowbase + 16 + l16) * DIM;
  const __bf16* b0p = Wm + (size_t)(colbase + l16)      * DIM;
  const __bf16* b1p = Wm + (size_t)(colbase + 16 + l16) * DIM;

  v8f c[2][2] = {};
  for (int kk = 0; kk < DIM; kk += 32) {
    v16bf a0 = load_a_frag(a0p, kk, grp);
    v16bf a1 = load_a_frag(a1p, kk, grp);
    v16bf b0 = load_b_frag(b0p, kk, grp);
    v16bf b1 = load_b_frag(b1p, kk, grp);
    c[0][0] = wmma_bf16(a0, b0, c[0][0]);
    c[0][1] = wmma_bf16(a0, b1, c[0][1]);
    c[1][0] = wmma_bf16(a1, b0, c[1][0]);
    c[1][1] = wmma_bf16(a1, b1, c[1][1]);
  }

  for (int nj = 0; nj < 2; nj++) {
    const int n = colbase + nj * 16 + l16;
    const float bv = bias[n];
    for (int mi = 0; mi < 2; mi++) {
      if (MODE == 0) {
        __bf16* Y = (__bf16*)Yv;
        for (int r = 0; r < 8; r++) {
          int m = rowbase + mi * 16 + r + 8 * grp;
          Y[(size_t)m * DIM + n] = (__bf16)(c[mi][nj][r] + bv);
        }
      } else if (MODE == 1) {
        __bf16* Y = (__bf16*)Yv;
        int m0 = rowbase + mi * 16 + 8 * grp;   // 8 consecutive tokens per lane
        int bidx = m0 / SEQ;
        int s0   = m0 % SEQ;
        int hh = n >> 6, dd = n & 63;
        v8bf pk;
        for (int r = 0; r < 8; r++) pk[r] = (__bf16)(c[mi][nj][r] + bv);
        *reinterpret_cast<v8bf*>(Y + (size_t)((bidx * NH + hh) * HD + dd) * SEQ + s0) = pk;
      } else {
        float* Y = (float*)Yv;
        for (int r = 0; r < 8; r++) {
          int m = rowbase + mi * 16 + r + 8 * grp;
          Y[(size_t)m * DIM + n] = c[mi][nj][r] + bv;
        }
      }
    }
  }
}

// ---------------- flash attention (causal), TDM-staged K/V, one wave = 32 q-rows ----------------
__global__ __launch_bounds__(128) void attn_kernel(const __bf16* __restrict__ Qp,
                                                   const __bf16* __restrict__ Kp,
                                                   const __bf16* __restrict__ Vt,
                                                   __bf16* __restrict__ ctx) {
  __shared__ __align__(16) __bf16 Kbuf[2][32 * 64];  // [key][dk] row-major, double-buffered
  __shared__ __align__(16) __bf16 Vbuf[2][64 * 32];  // [dk][key] row-major, double-buffered
  __shared__ __align__(16) __bf16 Pbuf[4][32 * 32];  // wave-private P tiles

  const int lane = threadIdx.x & 31;
  const int l16  = lane & 15;
  const int grp  = lane >> 4;
  const int w    = threadIdx.x >> 5;
  const int bh   = blockIdx.y;
  const int b    = bh >> 4;
  const int h    = bh & 15;
  const int qb0  = blockIdx.x * 128;
  const int qbase = qb0 + w * 32;

  const __bf16* Qb = Qp + (size_t)b * SEQ * DIM + h * HD;
  const __bf16* Kb = Kp + (size_t)b * SEQ * DIM + h * HD;
  const __bf16* Vb = Vt + (size_t)(b * NH + h) * HD * SEQ;
  __bf16* Pw = Pbuf[w];

  // Q fragments held in registers for the whole pass: [mi][kstep]
  v16bf qf[2][2];
  for (int mi = 0; mi < 2; mi++)
    for (int ks = 0; ks < 2; ks++)
      qf[mi][ks] = load_a_frag(Qb + (size_t)(qbase + mi * 16 + l16) * DIM, ks * 32, grp);

  v8f o[2][4] = {};
  float mstat[2][8], lstat[2][8];
  for (int mi = 0; mi < 2; mi++)
    for (int r = 0; r < 8; r++) { mstat[mi][r] = -1e30f; lstat[mi][r] = 0.f; }

  const int mytiles = (qbase >> 5) + 1;   // this wave's causal tile count
  const int ntiles  = (qb0 >> 5) + 4;     // block-wide tile count (all waves iterate)

  // prefetch tile 0 via Tensor Data Mover (wave 0 issues; EXEC ignored by TDM)
  if (w == 0) {
    tdm_load_2d(Kb, lds_off(Kbuf[0]), HD, 32, DIM);
    tdm_load_2d(Vb, lds_off(Vbuf[0]), 32, HD, SEQ);
  }

  for (int kt = 0; kt < ntiles; kt++) {
    const int kbase = kt * 32;
    const int cur = kt & 1;

    if (w == 0) __builtin_amdgcn_s_wait_tensorcnt(0);  // tile kt landed in LDS
    __syncthreads();                                    // visible to all; prev buf free
    if (w == 0 && kt + 1 < ntiles) {                    // start DMA of tile kt+1
      tdm_load_2d(Kb + (size_t)(kbase + 32) * DIM, lds_off(Kbuf[cur ^ 1]), HD, 32, DIM);
      tdm_load_2d(Vb + (kbase + 32), lds_off(Vbuf[cur ^ 1]), 32, HD, SEQ);
    }

    if (kt < mytiles) {   // wave-uniform predicate: EXEC stays all-ones inside
      const __bf16* KL = Kbuf[cur];
      const __bf16* VL = Vbuf[cur];
      const bool diag = (kt == mytiles - 1);  // mask needed only on the diagonal tile

      // scores S = Q K^T (32 q-rows x 32 keys), contraction over HD=64, K from LDS
      v8f sc[2][2] = {};
      for (int nj = 0; nj < 2; nj++) {
        const __bf16* krow = KL + (nj * 16 + l16) * HD;
        v16bf k0f = load_b_frag(krow, 0, grp);
        v16bf k1f = load_b_frag(krow, 32, grp);
        for (int mi = 0; mi < 2; mi++) {
          sc[mi][nj] = wmma_bf16(qf[mi][0], k0f, sc[mi][nj]);
          sc[mi][nj] = wmma_bf16(qf[mi][1], k1f, sc[mi][nj]);
        }
      }

      // online softmax (scale 1/sqrt(64) = 0.125, fp32)
      for (int mi = 0; mi < 2; mi++) {
        float rmax[8], alpha[8], rsum[8];
        for (int r = 0; r < 8; r++) {
          float v0 = sc[mi][0][r] * 0.125f;
          float v1 = sc[mi][1][r] * 0.125f;
          if (diag) {
            int mg = qbase + mi * 16 + r + 8 * grp;
            if (kbase + l16 > mg)      v0 = -1e30f;
            if (kbase + 16 + l16 > mg) v1 = -1e30f;
          }
          sc[mi][0][r] = v0; sc[mi][1][r] = v1;
          rmax[r] = fmaxf(v0, v1);
        }
        for (int d = 1; d < 16; d <<= 1)
          for (int r = 0; r < 8; r++)
            rmax[r] = fmaxf(rmax[r], __shfl_xor(rmax[r], d, 32));
        for (int r = 0; r < 8; r++) {
          float mn = fmaxf(mstat[mi][r], rmax[r]);
          alpha[r] = __expf(mstat[mi][r] - mn);
          mstat[mi][r] = mn;
        }
        for (int r = 0; r < 8; r++) {
          float p0 = __expf(sc[mi][0][r] - mstat[mi][r]);
          float p1 = __expf(sc[mi][1][r] - mstat[mi][r]);
          sc[mi][0][r] = p0; sc[mi][1][r] = p1;
          rsum[r] = p0 + p1;
        }
        for (int d = 1; d < 16; d <<= 1)
          for (int r = 0; r < 8; r++)
            rsum[r] += __shfl_xor(rsum[r], d, 32);
        for (int r = 0; r < 8; r++)
          lstat[mi][r] = lstat[mi][r] * alpha[r] + rsum[r];
        for (int nd = 0; nd < 4; nd++)
          for (int r = 0; r < 8; r++)
            o[mi][nd][r] *= alpha[r];
        // stage P (C-layout) into wave-private LDS tile, row-major [m][n], bf16
        for (int nj = 0; nj < 2; nj++)
          for (int r = 0; r < 8; r++)
            Pw[(mi * 16 + r + 8 * grp) * 32 + nj * 16 + l16] = (__bf16)sc[mi][nj][r];
      }

      // cross-lane LDS transpose: drain DS stores before A-layout reads
      asm volatile("s_wait_dscnt 0" ::: "memory");

      v16bf pa[2];
      for (int mi = 0; mi < 2; mi++) {
        U16 u;
        const __bf16* pr = Pw + (mi * 16 + l16) * 32;
        u.h[0] = *reinterpret_cast<const v8bf*>(pr + 8 * grp);
        u.h[1] = *reinterpret_cast<const v8bf*>(pr + 16 + 8 * grp);
        pa[mi] = u.v;
      }

      // O += P * V : B-fragments from LDS V tile (lane = dk col, keys contiguous)
      for (int nd = 0; nd < 4; nd++) {
        v16bf vf = load_b_frag(VL + (nd * 16 + l16) * 32, 0, grp);
        o[0][nd] = wmma_bf16(pa[0], vf, o[0][nd]);
        o[1][nd] = wmma_bf16(pa[1], vf, o[1][nd]);
      }
    }
  }

  // normalize and write context [B, SEQ, DIM]
  for (int mi = 0; mi < 2; mi++) {
    float inv[8];
    for (int r = 0; r < 8; r++) inv[r] = 1.0f / lstat[mi][r];
    for (int nd = 0; nd < 4; nd++)
      for (int r = 0; r < 8; r++) {
        int m = qbase + mi * 16 + r + 8 * grp;
        ctx[(size_t)(b * SEQ + m) * DIM + h * HD + nd * 16 + l16] =
            (__bf16)(o[mi][nd][r] * inv[r]);
      }
  }
}

extern "C" void kernel_launch(void* const* d_in, const int* in_sizes, int n_in,
                              void* d_out, int out_size, void* d_ws, size_t ws_size,
                              hipStream_t stream) {
  (void)in_sizes; (void)n_in; (void)out_size; (void)ws_size;

  const size_t ACT = (size_t)NB * SEQ * DIM;  // 4,194,304
  const size_t WSZ = (size_t)DIM * DIM;       // 1,048,576

  __bf16* qb  = reinterpret_cast<__bf16*>(d_ws);
  __bf16* kb  = qb  + ACT;
  __bf16* vb  = kb  + ACT;
  __bf16* wqb = vb  + ACT;
  __bf16* wkb = wqb + WSZ;
  __bf16* wvb = wkb + WSZ;
  __bf16* wob = wvb + WSZ;
  __bf16* Qp  = wob + WSZ;
  __bf16* Kp  = Qp  + ACT;
  __bf16* Vt  = Kp  + ACT;
  __bf16* ctx = Vt  + ACT;

  auto cvt = [&](int idx, __bf16* dst, size_t n) {
    int blocks = (int)((n / 4 + 255) / 256);
    cvt_bf16_kernel<<<blocks, 256, 0, stream>>>((const float*)d_in[idx], dst, (int)n);
  };
  cvt(0, qb, ACT);   // q
  cvt(1, kb, ACT);   // k
  cvt(2, vb, ACT);   // v
  cvt(4, wqb, WSZ);  // wq_w
  cvt(6, wkb, WSZ);  // wk_w
  cvt(8, wvb, WSZ);  // wv_w
  cvt(10, wob, WSZ); // wo_w

  dim3 pgrid(DIM / 64, (NB * SEQ) / 64);  // 128 threads = 4 waves, 32x32 tile per wave
  proj_kernel<0><<<pgrid, 128, 0, stream>>>(qb, wqb, (const float*)d_in[5], Qp);
  proj_kernel<0><<<pgrid, 128, 0, stream>>>(kb, wkb, (const float*)d_in[7], Kp);
  proj_kernel<1><<<pgrid, 128, 0, stream>>>(vb, wvb, (const float*)d_in[9], Vt);

  dim3 agrid(SEQ / 128, NB * NH);  // (16, 32)
  attn_kernel<<<agrid, 128, 0, stream>>>(Qp, Kp, Vt, ctx);

  proj_kernel<2><<<pgrid, 128, 0, stream>>>(ctx, wob, (const float*)d_in[11], d_out);
}
